// MultiScaleRetention_3281355014307
// MI455X (gfx1250) — matrix operand order repaired and checked by reading, not compiled, not run on hardware
//
#include <hip/hip_runtime.h>

// MultiScaleRetention for MI455X (gfx1250, wave32).
// bf16 WMMA GEMMs with async global->LDS double-buffered B tiles.
// B=4, S=1024, DIM=2048, H=16, HD=128.

typedef __attribute__((ext_vector_type(16))) __bf16 v16bf;
typedef __attribute__((ext_vector_type(8)))  __bf16 v8bf;
typedef __attribute__((ext_vector_type(8)))  float  v8f;

#define DEV __device__ __forceinline__

static constexpr int Bn = 4, S = 1024, DIM = 2048, H = 16, HD = 128;
static constexpr int BH = Bn * HD;          // 512 (batch-concat K dim)
static constexpr int M_TOK = Bn * S;        // 4096 tokens

// LDS B-tile: 64 rows x 32 K bf16, row stride 32 elems; 2 buffers.
static constexpr int BTILE_ELEMS = 64 * 32;         // 2048 bf16 = 4KB

// ---------------- WMMA fragment helpers (wave32, 16x16x32 bf16) -------------
// A 16x32 (16-bit): lanes 0-15 -> M=lane, K in {0..7,16..23}; lanes 16-31 ->
// M=lane-16, K in {8..15,24..31}.
DEV v16bf load_a_f32(const float* p0, int ld, int lane) {
  int m   = lane & 15;
  int klo = (lane & 16) ? 8 : 0;
  const float* p = p0 + (size_t)m * ld + klo;
  float4 a0 = *(const float4*)(p);
  float4 a1 = *(const float4*)(p + 4);
  float4 a2 = *(const float4*)(p + 16);
  float4 a3 = *(const float4*)(p + 20);
  v16bf r;
  r[0]=(__bf16)a0.x; r[1]=(__bf16)a0.y; r[2]=(__bf16)a0.z; r[3]=(__bf16)a0.w;
  r[4]=(__bf16)a1.x; r[5]=(__bf16)a1.y; r[6]=(__bf16)a1.z; r[7]=(__bf16)a1.w;
  r[8]=(__bf16)a2.x; r[9]=(__bf16)a2.y; r[10]=(__bf16)a2.z; r[11]=(__bf16)a2.w;
  r[12]=(__bf16)a3.x; r[13]=(__bf16)a3.y; r[14]=(__bf16)a3.z; r[15]=(__bf16)a3.w;
  return r;
}

DEV v16bf load_a_bf16(const __bf16* p0, int ld, int lane) {
  int m   = lane & 15;
  int klo = (lane & 16) ? 8 : 0;
  const __bf16* p = p0 + (size_t)m * ld + klo;
  v8bf lo = *(const v8bf*)(p);        // K = klo..klo+7
  v8bf hi = *(const v8bf*)(p + 16);   // K = 16+klo..16+klo+7
  v16bf r;
#pragma unroll
  for (int i = 0; i < 8; ++i) { r[i] = lo[i]; r[8 + i] = hi[i]; }
  return r;
}

// B 32x16 fragment read from LDS tile (rows = N, 32 K elems contiguous):
// lanes 0-15 -> N=lane, K in {0..7,16..23}; lanes 16-31 -> N=lane-16, K in {8..15,24..31}.
DEV v16bf load_b_lds(const __bf16* base, int lane) {
  int n   = lane & 15;
  int klo = (lane & 16) ? 8 : 0;
  const __bf16* p = base + n * 32 + klo;
  v8bf lo = *(const v8bf*)(p);
  v8bf hi = *(const v8bf*)(p + 16);
  v16bf r;
#pragma unroll
  for (int i = 0; i < 8; ++i) { r[i] = lo[i]; r[8 + i] = hi[i]; }
  return r;
}

DEV v8f wmma_bf16(v16bf a, v16bf b, v8f c) {
  return __builtin_amdgcn_wmma_f32_16x16x32_bf16(false, a, false, b,
                                                 (short)0, c, false, false);
}

DEV v8f vzero8() { v8f z = {0.f,0.f,0.f,0.f,0.f,0.f,0.f,0.f}; return z; }

// Issue async copy of one B tile K-step into LDS buffer (4KB, 256 x b128).
// 128 threads, 2 chunks each.  Tracked by ASYNCcnt.
DEV void issue_btile(const __bf16* Bt, int ldb, int k,
                     unsigned lds_base, int tid) {
#pragma unroll
  for (int it = 0; it < 2; ++it) {
    int c  = tid + it * 128;            // chunk 0..255
    int r  = c >> 2;                    // tile row (N) 0..63
    int co = (c & 3) * 8;               // elem offset in row (8 bf16 = 16B)
    unsigned long long g =
        (unsigned long long)(uintptr_t)(Bt + (size_t)r * ldb + k + co);
    unsigned lds = lds_base + (unsigned)(r * 64 + co * 2);  // bytes
    asm volatile("global_load_async_to_lds_b128 %0, %1, off"
                 :: "v"(lds), "v"(g) : "memory");
  }
}

// ---------------- shared GEMM pipeline --------------------------------------
// Block = 128 threads (4 waves).  Block tile 64(M) x 64(N).  Wave w owns rows
// [w*16, w*16+16).  B tile staged in LDS via async loads, double-buffered.
// Abase: per-wave pointer to its 16 rows (f32 or bf16).  Bt: (N x K) bf16,
// pre-offset to col0.  acc[4]: 16x16 tiles over N = col0 + j*16.
template<bool AF32>
DEV void gemm_pipe(const void* Abase, int lda,
                   const __bf16* Bt, int ldb, int K,
                   v8f acc[4], __bf16* shb) {
  int tid  = threadIdx.x;
  int lane = tid & 31;
  unsigned shb_base = (unsigned)(uintptr_t)shb;   // LDS byte offset (low 32b)

  issue_btile(Bt, ldb, 0, shb_base, tid);
  int nsteps = K >> 5;
  for (int i = 0; i < nsteps; ++i) {
    int k = i << 5;
    if (i + 1 < nsteps) {
      // prefetch next K-step into the other buffer (safe: trailing barrier of
      // step i-1 guaranteed everyone finished reading that buffer)
      issue_btile(Bt, ldb, k + 32, shb_base + ((i + 1) & 1) * 4096, tid);
      asm volatile("s_wait_asynccnt 0x2" ::: "memory");  // buf i landed
    } else {
      asm volatile("s_wait_asynccnt 0x0" ::: "memory");
    }
    __syncthreads();                                    // visible to all waves

    const __bf16* bufp = shb + (i & 1) * BTILE_ELEMS;
    v16bf a;
    if constexpr (AF32) a = load_a_f32((const float*)Abase + k, lda, lane);
    else                a = load_a_bf16((const __bf16*)Abase + k, lda, lane);
#pragma unroll
    for (int j = 0; j < 4; ++j) {
      v16bf b = load_b_lds(bufp + j * 16 * 32, lane);
      acc[j] = wmma_bf16(a, b, acc[j]);
    }
    __syncthreads();               // done reading buf i -> reusable at step i+2
  }
}

// ---------------- conversion / transpose kernels ----------------------------
__global__ __launch_bounds__(256) void k_conv_wht(const float* __restrict__ Wh,
                                                  __bf16* __restrict__ wht) {
  int i = blockIdx.x * 256 + threadIdx.x;        // H*HD*HD = 262144
  int h = i >> 14, rem = i & 16383, e = rem >> 7, d = rem & 127;
  wht[i] = (__bf16)Wh[((size_t)(h * HD + d)) * HD + e];   // wht[h][e][d]
}

__global__ __launch_bounds__(256) void k_convT2048(const float* __restrict__ Win,
                                                   __bf16* __restrict__ Wt) {
  int i = blockIdx.x * 256 + threadIdx.x;        // DIM*DIM, i = n*2048 + k
  int n = i >> 11, k = i & 2047;
  Wt[i] = (__bf16)Win[((size_t)k << 11) + n];    // Wt[n][k] = W[k][n]
}

// ---------------- shared-weight per-head projection (Q,K,V) -----------------
__global__ __launch_bounds__(128) void k_proj(const float* __restrict__ src,
                                              const __bf16* __restrict__ wht,
                                              __bf16* __restrict__ dst,
                                              int transposed_v) {
  __shared__ __bf16 shb[2 * BTILE_ELEMS];
  int lane = threadIdx.x & 31, wave = threadIdx.x >> 5;
  int h = blockIdx.z;
  int row0 = blockIdx.y * 64 + wave * 16;
  int col0 = blockIdx.x * 64;
  const float*  A  = src + (size_t)row0 * DIM + h * HD;
  const __bf16* Bt = wht + (size_t)h * HD * HD + (size_t)col0 * HD;

  v8f acc[4];
#pragma unroll
  for (int j = 0; j < 4; ++j) acc[j] = vzero8();
  gemm_pipe<true>(A, DIM, Bt, HD, HD, acc, shb);

  int mbase = row0 + ((lane & 16) ? 8 : 0);
  int nlan  = lane & 15;
#pragma unroll
  for (int j = 0; j < 4; ++j) {
    int e = col0 + j * 16 + nlan;
#pragma unroll
    for (int r = 0; r < 8; ++r) {
      int m = mbase + r, b = m >> 10, si = m & 1023;
      __bf16 v = (__bf16)acc[j][r];
      if (!transposed_v)   // q/k:  [h][s][b*HD+e]
        dst[((size_t)(h * S + si)) * BH + b * HD + e] = v;
      else                 // v:    [h][b*HD+e][t]
        dst[((size_t)(h * BH + b * HD + e)) * S + si] = v;
    }
  }
}

// ---------------- batch-summed scores + decay mask + clamp ------------------
__global__ __launch_bounds__(128) void k_scores(const __bf16* __restrict__ qh,
                                                const __bf16* __restrict__ kh,
                                                __bf16* __restrict__ Amat) {
  __shared__ __bf16 shb[2 * BTILE_ELEMS];
  int lane = threadIdx.x & 31, wave = threadIdx.x >> 5;
  int h = blockIdx.z;
  int row0 = blockIdx.y * 64 + wave * 16;   // s
  int col0 = blockIdx.x * 64;               // t
  const __bf16* A  = qh + ((size_t)h * S + row0) * BH;
  const __bf16* Bt = kh + ((size_t)h * S + col0) * BH;

  v8f acc[4];
#pragma unroll
  for (int j = 0; j < 4; ++j) acc[j] = vzero8();
  gemm_pipe<false>(A, BH, Bt, BH, BH, acc, shb);

  const float inv_sqrt_hd = 0.08838834764831845f;     // 1/sqrt(128)
  float gam  = 1.0f - exp2f(-(float)(5 + h));
  float cgam = gam / (1.0f - gam);
  int sbase = row0 + ((lane & 16) ? 8 : 0);
  int nlan  = lane & 15;
#pragma unroll
  for (int j = 0; j < 4; ++j) {
    int t = col0 + j * 16 + nlan;
    // rowsum[h][t] = sum_{m=0..t} gam^{-m} = (gam^{-(t+1)} - 1) * gam/(1-gam)
    float rowsum = (__powf(gam, -(float)(t + 1)) - 1.0f) * cgam;
    float rs = rsqrtf(rowsum);
#pragma unroll
    for (int r = 0; r < 8; ++r) {
      int s = sbase + r;
      float dn = (t <= s) ? __powf(gam, (float)(t - s)) * rs : 0.0f;
      float v  = fabsf(acc[j][r] * inv_sqrt_hd * dn);
      v = fmaxf(v, 1.0f);                 // upper triangle -> 1.0 (dense!)
      Amat[((size_t)h * S + s) * S + t] = (__bf16)v;
    }
  }
}

// ---------------- O = A[h] (SxS) @ Vcat (S x 512) ---------------------------
__global__ __launch_bounds__(128) void k_ogemm(const __bf16* __restrict__ Amat,
                                               const __bf16* __restrict__ vt,
                                               float* __restrict__ O) {
  __shared__ __bf16 shb[2 * BTILE_ELEMS];
  int lane = threadIdx.x & 31, wave = threadIdx.x >> 5;
  int h = blockIdx.z;
  int row0 = blockIdx.y * 64 + wave * 16;   // s
  int col0 = blockIdx.x * 64;               // n = b*HD+e
  const __bf16* A  = Amat + ((size_t)h * S + row0) * S;
  const __bf16* Bt = vt   + ((size_t)h * BH + col0) * S;

  v8f acc[4];
#pragma unroll
  for (int j = 0; j < 4; ++j) acc[j] = vzero8();
  gemm_pipe<false>(A, S, Bt, S, S, acc, shb);

  int sbase = row0 + ((lane & 16) ? 8 : 0);
  int nlan  = lane & 15;
#pragma unroll
  for (int j = 0; j < 4; ++j) {
    int n = col0 + j * 16 + nlan, b = n >> 7, e = n & 127;
#pragma unroll
    for (int r = 0; r < 8; ++r) {
      int s = sbase + r;
      O[((size_t)(b * S + s)) * DIM + h * HD + e] = acc[j][r];
    }
  }
}

// ---------------- per-token layernorm stats (GroupNorm groups=1) ------------
__global__ __launch_bounds__(256) void k_gnstats(const float* __restrict__ O,
                                                 float* __restrict__ mu,
                                                 float* __restrict__ rstd) {
  int tok = blockIdx.x;
  const float* row = O + (size_t)tok * DIM;
  float s = 0.f, sq = 0.f;
  for (int i = threadIdx.x; i < DIM; i += 256) {
    float v = row[i]; s += v; sq += v * v;
  }
  __shared__ float sh1[256], sh2[256];
  sh1[threadIdx.x] = s; sh2[threadIdx.x] = sq; __syncthreads();
  for (int o = 128; o > 0; o >>= 1) {
    if (threadIdx.x < o) {
      sh1[threadIdx.x] += sh1[threadIdx.x + o];
      sh2[threadIdx.x] += sh2[threadIdx.x + o];
    }
    __syncthreads();
  }
  if (threadIdx.x == 0) {
    float m = sh1[0] * (1.0f / DIM);
    float var = sh2[0] * (1.0f / DIM) - m * m;
    mu[tok] = m; rstd[tok] = rsqrtf(var + 1e-3f);
  }
}

// ---------------- gate GEMM  G = relu(x@wg) * GN(O) -------------------------
__global__ __launch_bounds__(128) void k_gate(const float* __restrict__ x,
                                              const __bf16* __restrict__ wgT,
                                              const float* __restrict__ O,
                                              const float* __restrict__ mu,
                                              const float* __restrict__ rstd,
                                              const float* __restrict__ gnw,
                                              const float* __restrict__ gnb,
                                              __bf16* __restrict__ G) {
  __shared__ __bf16 shb[2 * BTILE_ELEMS];
  int lane = threadIdx.x & 31, wave = threadIdx.x >> 5;
  int row0 = blockIdx.y * 64 + wave * 16;
  int col0 = blockIdx.x * 64;
  const float*  A  = x   + (size_t)row0 * DIM;
  const __bf16* Bt = wgT + (size_t)col0 * DIM;

  v8f acc[4];
#pragma unroll
  for (int j = 0; j < 4; ++j) acc[j] = vzero8();
  gemm_pipe<true>(A, DIM, Bt, DIM, DIM, acc, shb);

  int mbase = row0 + ((lane & 16) ? 8 : 0);
  int nlan  = lane & 15;
#pragma unroll
  for (int j = 0; j < 4; ++j) {
    int jc = col0 + j * 16 + nlan;
#pragma unroll
    for (int r = 0; r < 8; ++r) {
      int m = mbase + r;
      float w = fmaxf(acc[j][r], 0.0f);
      float y = (O[(size_t)m * DIM + jc] - mu[m]) * rstd[m] * gnw[jc] + gnb[jc];
      G[(size_t)m * DIM + jc] = (__bf16)(w * y);
    }
  }
}

// ---------------- output GEMM  out = G @ wo ---------------------------------
__global__ __launch_bounds__(128) void k_out(const __bf16* __restrict__ G,
                                             const __bf16* __restrict__ woT,
                                             float* __restrict__ out) {
  __shared__ __bf16 shb[2 * BTILE_ELEMS];
  int lane = threadIdx.x & 31, wave = threadIdx.x >> 5;
  int row0 = blockIdx.y * 64 + wave * 16;
  int col0 = blockIdx.x * 64;
  const __bf16* A  = G   + (size_t)row0 * DIM;
  const __bf16* Bt = woT + (size_t)col0 * DIM;

  v8f acc[4];
#pragma unroll
  for (int j = 0; j < 4; ++j) acc[j] = vzero8();
  gemm_pipe<false>(A, DIM, Bt, DIM, DIM, acc, shb);

  int mbase = row0 + ((lane & 16) ? 8 : 0);
  int nlan  = lane & 15;
#pragma unroll
  for (int j = 0; j < 4; ++j) {
    int n = col0 + j * 16 + nlan;
#pragma unroll
    for (int r = 0; r < 8; ++r)
      out[(size_t)(mbase + r) * DIM + n] = acc[j][r];
  }
}

// ---------------------------------------------------------------------------
extern "C" void kernel_launch(void* const* d_in, const int* in_sizes, int n_in,
                              void* d_out, int out_size, void* d_ws, size_t ws_size,
                              hipStream_t stream) {
  const float* x   = (const float*)d_in[0];
  const float* kin = (const float*)d_in[1];
  const float* vin = (const float*)d_in[2];
  const float* Wh  = (const float*)d_in[3];
  const float* wg  = (const float*)d_in[4];
  const float* wo  = (const float*)d_in[5];
  const float* gnw = (const float*)d_in[6];
  const float* gnb = (const float*)d_in[7];
  float* out = (float*)d_out;

  char* ws = (char*)d_ws;
  size_t off = 0;
  auto alloc = [&](size_t bytes) -> void* {
    void* p = ws + off; off += (bytes + 255) & ~(size_t)255; return p;
  };
  __bf16* wht  = (__bf16*)alloc((size_t)H * HD * HD * 2);      // 0.5 MB
  __bf16* wgT  = (__bf16*)alloc((size_t)DIM * DIM * 2);        // 8.4 MB
  __bf16* woT  = (__bf16*)alloc((size_t)DIM * DIM * 2);        // 8.4 MB
  __bf16* qh   = (__bf16*)alloc((size_t)H * S * BH * 2);       // 16.8 MB
  __bf16* kh   = (__bf16*)alloc((size_t)H * S * BH * 2);       // 16.8 MB
  __bf16* vt   = (__bf16*)alloc((size_t)H * BH * S * 2);       // 16.8 MB
  __bf16* Amat = (__bf16*)alloc((size_t)H * S * S * 2);        // 33.6 MB
  float*  O    = (float*) alloc((size_t)M_TOK * DIM * 4);      // 33.6 MB
  float*  mu   = (float*) alloc((size_t)M_TOK * 4);
  float*  rstd = (float*) alloc((size_t)M_TOK * 4);
  __bf16* G    = Amat;   // alias: Amat dead after k_ogemm    (~135 MB total)

  dim3 blk(128);

  // weight conversions / transposes
  k_conv_wht <<<(H * HD * HD) / 256, 256, 0, stream>>>(Wh, wht);
  k_convT2048<<<(DIM * DIM) / 256,  256, 0, stream>>>(wg, wgT);
  k_convT2048<<<(DIM * DIM) / 256,  256, 0, stream>>>(wo, woT);

  // shared-weight projections (Q,K normal layout; V transposed per head)
  dim3 gproj(HD / 64, M_TOK / 64, H);
  k_proj<<<gproj, blk, 0, stream>>>(x,   wht, qh, 0);
  k_proj<<<gproj, blk, 0, stream>>>(kin, wht, kh, 0);
  k_proj<<<gproj, blk, 0, stream>>>(vin, wht, vt, 1);

  // batch-summed scores with decay mask + abs/clamp -> dense A (bf16)
  dim3 gsc(S / 64, S / 64, H);
  k_scores<<<gsc, blk, 0, stream>>>(qh, kh, Amat);

  // O = A @ Vcat  -> (B,S,DIM) f32
  dim3 gog(BH / 64, S / 64, H);
  k_ogemm<<<gog, blk, 0, stream>>>(Amat, vt, O);

  // per-token layernorm stats
  k_gnstats<<<M_TOK, 256, 0, stream>>>(O, mu, rstd);

  // G = relu(x@wg) * GN(O)   (bf16), then out = G @ wo  (f32)
  dim3 gmm(DIM / 64, M_TOK / 64);
  k_gate<<<gmm, blk, 0, stream>>>(x, wgT, O, mu, rstd, gnw, gnb, G);
  k_out <<<gmm, blk, 0, stream>>>(G, woT, out);
}
